// ProtoNet_4483945857218
// MI455X (gfx1250) — matrix operand order repaired and checked by reading, not compiled, not run on hardware
//
#include <hip/hip_runtime.h>
#include <math.h>

// Problem constants (from reference setup)
#define NTOT 100          // total samples (75 queries + 25 supports, reordered)
#define NQRY 75
#define NSUP 25           // m ranges over 25 support samples
#define NCLS 5
#define CCH  128
#define TT   8
#define VV   25
#define TOKS_PER (TT*VV)  // 200 tokens per sample
#define FEAT (CCH*TT*VV)  // 25600 floats per sample
#define TILE_ELEMS (CCH*32)  // 4096 bf16 per (sample,t) K^T / V tile
#define LN_EPS 1e-12f

typedef __attribute__((ext_vector_type(16))) __bf16 bf16x16;
typedef __attribute__((ext_vector_type(8)))  float  f32x8;
typedef unsigned short u16;

// ---------- helpers ----------

__device__ __forceinline__ u16 f2bf(float f) {           // RNE f32 -> bf16 bits
  unsigned int u = __float_as_uint(f);
  u += 0x7FFFu + ((u >> 16) & 1u);
  return (u16)(u >> 16);
}

__device__ __forceinline__ f32x8 wmma_bf16(bf16x16 a, bf16x16 b, f32x8 c) {
  return __builtin_amdgcn_wmma_f32_16x16x32_bf16(false, a, false, b, (short)0, c, false, false);
}

// A fragment (16x32 MxK bf16) from a natural row-major tile.
// Layout: lanes0-15: K=0..15 ; lanes16-31: K=16..31 (matches B fragment order).
// Per lane this is one contiguous 32B run -> compiler emits ds_load_b128 pairs.
__device__ __forceinline__ bf16x16 load_a_frag(const u16* base, int rowBase, int rowStride,
                                               int kBase, int lane) {
  union { bf16x16 v; u16 u[16]; } f;
  const int hi = lane >> 4;
  const u16* rp = base + (rowBase + (lane & 15)) * rowStride + kBase;
#pragma unroll
  for (int e = 0; e < 16; ++e) {
    int k = (hi << 4) + e;
    f.u[e] = rp[k];
  }
  return f.v;
}

// B fragment from fragment-major ("swizzled") storage: [frag][lane][e] with the
// 16 bf16 of each lane contiguous (32B) -> two b128 loads.
// Element e of lane maps to B[k = kTile*32 + e + 16*(lane>>4)][n = nTile*16 + (lane&15)].
__device__ __forceinline__ bf16x16 load_frag_sw(const u16* base, int fragIdx, int lane) {
  union { bf16x16 v; uint4 q[2]; } f;
  const uint4* p = (const uint4*)(base + ((size_t)(fragIdx * 32 + lane) << 4));
  f.q[0] = p[0];
  f.q[1] = p[1];
  return f.v;
}

__device__ __forceinline__ float wave_sum(float x) {
#pragma unroll
  for (int off = 16; off > 0; off >>= 1) x += __shfl_xor(x, off, 32);
  return x;
}
__device__ __forceinline__ float wave_max(float x) {
#pragma unroll
  for (int off = 16; off > 0; off >>= 1) x = fmaxf(x, __shfl_xor(x, off, 32));
  return x;
}

// cat row n -> original sample index
__device__ __forceinline__ int perm_orig(int n) {
  if (n < NQRY) { int cl = n / 15; int j = n - cl * 15; return cl * 20 + 5 + j; }
  int s = n - NQRY; int cl = s / 5; int j = s - cl * 5;  return cl * 20 + j;
}

// ---------- stage 0: weights -> bf16, pre-swizzled fragment-major ----------

__global__ void k_weights(const float* __restrict__ Wq, const float* __restrict__ Wk,
                          const float* __restrict__ Wv, const float* __restrict__ Wo,
                          u16* __restrict__ dst) {
  int idx = blockIdx.x * blockDim.x + threadIdx.x;
  if (idx >= 4 * CCH * CCH) return;
  int which = idx >> 14, j = idx & 16383;
  const float* src = (which == 0) ? Wq : (which == 1) ? Wk : (which == 2) ? Wv : Wo;
  // dst layout per weight: frag = kk*8 + ni ; then [lane][e]
  int e    = j & 15;
  int lane = (j >> 4) & 31;
  int frag = j >> 9;                 // 0..31
  int kk = frag >> 3, ni = frag & 7;
  int hi = lane >> 4;
  int k  = kk * 32 + e + 16 * hi;    // input channel (B row / contraction dim)
  int n  = ni * 16 + (lane & 15);    // output channel
  dst[idx] = f2bf(src[k * CCH + n]);
}

// cat[n][t][v][c] = input[perm(n)][c][t][v]
__global__ void k_cat(const float* __restrict__ inp, float* __restrict__ cat,
                      u16* __restrict__ catBf) {
  int idx = blockIdx.x * blockDim.x + threadIdx.x;   // exactly NTOT*FEAT threads
  int c = idx & 127;
  int v = (idx >> 7) % VV;
  int t = (idx / (CCH * VV)) % TT;
  int n = idx / (CCH * VV * TT);
  float x = inp[((perm_orig(n) * CCH + c) * TT + t) * VV + v];
  cat[idx] = x;
  catBf[idx] = f2bf(x);
}

// ---------- stage 1: token projections via WMMA ----------
// mode 0: out[tok][cout]              (Q, natural A layout for the scores GEMM)
// mode 1: K^T fragment-major tiles    [s*TT+t][frag=kk*2+ni][lane][e], zero-padded
// mode 2: V   fragment-major tiles    [s*TT+t][frag=ni][lane][e],      zero-padded

__global__ void __launch_bounds__(256) k_proj(const u16* __restrict__ X,
                                              const u16* __restrict__ Wbf,
                                              const float* __restrict__ bias,
                                              u16* __restrict__ out,
                                              int nTok, int mode) {
  __shared__ u16 sA[32 * CCH];
  const int tid = threadIdx.x, lane = tid & 31, w = tid >> 5;
  const int tokBase = blockIdx.x * 32;

  // stage 32 token rows (vectorized, wave-uniform row predicate)
  {
    int r = tid >> 3;                    // 8 threads per 128-wide row
    int cseg = (tid & 7) * 16;
    uint4* d = (uint4*)(sA + r * CCH + cseg);
    int tok = tokBase + r;
    if (tok < nTok) {
      const uint4* s = (const uint4*)(X + (size_t)tok * CCH + cseg);
      d[0] = s[0]; d[1] = s[1];
    } else {
      uint4 z = {0u, 0u, 0u, 0u};
      d[0] = z; d[1] = z;
    }
  }
  __syncthreads();

  f32x8 acc0 = {}; f32x8 acc1 = {};
#pragma unroll
  for (int kk = 0; kk < 4; ++kk) {       // K = 128 in 4 steps of 32
    bf16x16 bfr = load_frag_sw(Wbf, kk * 8 + w, lane);
    bf16x16 a0  = load_a_frag(sA, 0,  CCH, kk * 32, lane);
    bf16x16 a1  = load_a_frag(sA, 16, CCH, kk * 32, lane);
    acc0 = wmma_bf16(a0, bfr, acc0);
    acc1 = wmma_bf16(a1, bfr, acc1);
  }

  const int hi = lane >> 4;
  const int cout = w * 16 + (lane & 15);
  const float badd = bias[cout];
#pragma unroll
  for (int mi = 0; mi < 2; ++mi) {
    f32x8 acc = mi ? acc1 : acc0;
#pragma unroll
    for (int r = 0; r < 8; ++r) {
      int M = mi * 16 + r + 8 * hi;
      int tok = tokBase + M;
      if (tok < nTok) {
        u16 val = f2bf(acc[r] + badd);
        if (mode == 0) {
          out[(size_t)tok * CCH + cout] = val;
        } else {
          int s = tok / TOKS_PER;
          int rem = tok - s * TOKS_PER;
          int t = rem / VV, v = rem - t * VV;
          size_t blk = (size_t)(s * TT + t) * TILE_ELEMS;
          if (mode == 1) {                       // K^T: contraction dim = cout
            int kk2 = cout >> 5, r32 = cout & 31;
            int hi2 = r32 >> 4, e2 = r32 & 15;
            int ni = v >> 4;
            int lane2 = hi2 * 16 + (v & 15);
            out[blk + (size_t)(((kk2 * 2 + ni) * 32 + lane2) << 4) + e2] = val;
            if (v == 0) {                        // bake zero pad for v'=25..31
#pragma unroll
              for (int vp = VV; vp < 32; ++vp)
                out[blk + (size_t)(((kk2 * 2 + 1) * 32 + hi2 * 16 + (vp & 15)) << 4) + e2] = 0;
            }
          } else {                               // V: contraction dim = v
            int hi2 = v >> 4, e2 = v & 15;
            int ni = cout >> 4;
            int lane2 = hi2 * 16 + (cout & 15);
            out[blk + (size_t)((ni * 32 + lane2) << 4) + e2] = val;
            if (v == 0) {                        // pad rows k=25..31
#pragma unroll
              for (int vp = VV; vp < 32; ++vp)
                out[blk + (size_t)((ni * 32 + 16 + (cout & 15)) << 4) + (vp & 15)] = 0;
            }
          }
        }
      }
    }
  }
}

// ---------- stage 2: attention + Wo + LayerNorm + mean over m ----------
// one workgroup (8 wave32) per (n,t); loops m = 0..24 entirely in LDS.

__global__ void __launch_bounds__(256) k_attn(const u16* __restrict__ Qbf,
                                              const u16* __restrict__ Ktb,
                                              const u16* __restrict__ Vbf,
                                              const u16* __restrict__ Wob,
                                              const float* __restrict__ cat,
                                              const float* __restrict__ bo,
                                              const float* __restrict__ ln_g,
                                              const float* __restrict__ ln_b,
                                              float* __restrict__ att) {
  __shared__ u16   sQ[32 * CCH];       // [q][c] natural (A source)
  __shared__ u16   sK[TILE_ELEMS];     // K^T fragment-major (B source)
  __shared__ u16   sV[TILE_ELEMS];     // V fragment-major (B source)
  __shared__ u16   sAttn[32 * 32];     // softmax probs bf16 (zero padded)
  __shared__ u16   sCtx[32 * CCH];     // ctx bf16 natural (A source)
  __shared__ float sY[32 * CCH];       // Wo GEMM out f32 (aliases scores)
  __shared__ float sAcc[VV * CCH];     // running sum over m of LN outputs

  const int tid = threadIdx.x, lane = tid & 31, w = tid >> 5;
  const int hi = lane >> 4, ln15 = lane & 15;
  const int n = blockIdx.x >> 3, t = blockIdx.x & 7;
  const int qBase = ((n * TT + t) * VV) * CCH;
  const float scale = 0.08838834764831845f;     // 1/sqrt(128)
  float* sS = sY;                               // scores alias (32x32 f32)

  for (int i = tid; i < VV * CCH; i += 256) sAcc[i] = 0.f;

  // stage Q tile once (vectorized)
  {
    int q = tid >> 3;
    int cseg = (tid & 7) * 16;
    uint4* d = (uint4*)(sQ + q * CCH + cseg);
    if (q < VV) {
      const uint4* s = (const uint4*)(Qbf + qBase + q * CCH + cseg);
      d[0] = s[0]; d[1] = s[1];
    } else {
      uint4 z = {0u, 0u, 0u, 0u};
      d[0] = z; d[1] = z;
    }
  }

  // Wo B fragments are m-invariant: keep them in registers for the whole loop
  bf16x16 wof0 = load_frag_sw(Wob, 0 * 8 + w, lane);
  bf16x16 wof1 = load_frag_sw(Wob, 1 * 8 + w, lane);
  bf16x16 wof2 = load_frag_sw(Wob, 2 * 8 + w, lane);
  bf16x16 wof3 = load_frag_sw(Wob, 3 * 8 + w, lane);

  for (int m = 0; m < NSUP; ++m) {
    __syncthreads();                            // prev-iter LN done before reuse
    // stage K^T / V fragment-major tiles: straight vector copies (pads baked in)
    {
      const u16* kBlk = Ktb + (size_t)(m * TT + t) * TILE_ELEMS;
      const u16* vBlk = Vbf + (size_t)(m * TT + t) * TILE_ELEMS;
      const uint4* ks = (const uint4*)(kBlk + tid * 16);
      const uint4* vs = (const uint4*)(vBlk + tid * 16);
      uint4* kd = (uint4*)(sK + tid * 16);
      uint4* vd = (uint4*)(sV + tid * 16);
      kd[0] = ks[0]; kd[1] = ks[1];
      vd[0] = vs[0]; vd[1] = vs[1];
      if (m + 1 < NSUP) {                       // pull next tiles toward the WGP
        __builtin_prefetch(Ktb + (size_t)((m + 1) * TT + t) * TILE_ELEMS + tid * 16, 0, 0);
        __builtin_prefetch(Vbf + (size_t)((m + 1) * TT + t) * TILE_ELEMS + tid * 16, 0, 0);
      }
    }
    __syncthreads();

    // scores(25x25 pad 32x32) = Q @ K^T, waves 0..3 (wave-uniform branch)
    if (w < 4) {
      int mi = w >> 1, ni = w & 1;
      f32x8 acc = {};
#pragma unroll
      for (int kk = 0; kk < 4; ++kk) {
        bf16x16 a = load_a_frag(sQ, mi * 16, CCH, kk * 32, lane);
        bf16x16 b = load_frag_sw(sK, kk * 2 + ni, lane);
        acc = wmma_bf16(a, b, acc);
      }
#pragma unroll
      for (int r = 0; r < 8; ++r)
        sS[(mi * 16 + r + 8 * hi) * 32 + ni * 16 + ln15] = acc[r] * scale;
    }
    __syncthreads();

    // row softmax (wave per row), write bf16 probs, zero pad rows/cols
#pragma unroll
    for (int i = 0; i < 4; ++i) {
      int r = w + 8 * i;
      if (r < VV) {
        float s = (lane < VV) ? sS[r * 32 + lane] : -INFINITY;
        float mx = wave_max(s);
        float p = (lane < VV) ? __expf(s - mx) : 0.f;
        float sum = wave_sum(p);
        sAttn[r * 32 + lane] = f2bf(p / sum);
      } else {
        sAttn[r * 32 + lane] = 0;
      }
    }
    __syncthreads();

    // ctx(25x128) = attn @ V, K=32 (zero-padded) ; B frag hoisted over mi
    {
      bf16x16 bv = load_frag_sw(sV, w, lane);
#pragma unroll
      for (int mi = 0; mi < 2; ++mi) {
        bf16x16 a = load_a_frag(sAttn, mi * 16, 32, 0, lane);
        f32x8 acc = {};
        acc = wmma_bf16(a, bv, acc);
#pragma unroll
        for (int r = 0; r < 8; ++r)
          sCtx[(mi * 16 + r + 8 * hi) * CCH + w * 16 + ln15] = f2bf(acc[r]);
      }
    }
    __syncthreads();

    // y(25x128) = ctx @ Wo (B fragments live in registers)
    {
      f32x8 acc0 = {}; f32x8 acc1 = {};
      bf16x16 a0, a1;
      a0 = load_a_frag(sCtx, 0,  CCH, 0, lane);
      a1 = load_a_frag(sCtx, 16, CCH, 0, lane);
      acc0 = wmma_bf16(a0, wof0, acc0);  acc1 = wmma_bf16(a1, wof0, acc1);
      a0 = load_a_frag(sCtx, 0,  CCH, 32, lane);
      a1 = load_a_frag(sCtx, 16, CCH, 32, lane);
      acc0 = wmma_bf16(a0, wof1, acc0);  acc1 = wmma_bf16(a1, wof1, acc1);
      a0 = load_a_frag(sCtx, 0,  CCH, 64, lane);
      a1 = load_a_frag(sCtx, 16, CCH, 64, lane);
      acc0 = wmma_bf16(a0, wof2, acc0);  acc1 = wmma_bf16(a1, wof2, acc1);
      a0 = load_a_frag(sCtx, 0,  CCH, 96, lane);
      a1 = load_a_frag(sCtx, 16, CCH, 96, lane);
      acc0 = wmma_bf16(a0, wof3, acc0);  acc1 = wmma_bf16(a1, wof3, acc1);
#pragma unroll
      for (int r = 0; r < 8; ++r) {
        sY[(r + 8 * hi) * CCH + w * 16 + ln15]      = acc0[r];
        sY[(16 + r + 8 * hi) * CCH + w * 16 + ln15] = acc1[r];
      }
    }
    __syncthreads();

    // LayerNorm(y + bo + residual) over c, accumulate mean over m (wave per row)
#pragma unroll
    for (int i = 0; i < 4; ++i) {
      int r = w + 8 * i;
      if (r < VV) {
        float v0 = sY[r * CCH + lane]      + bo[lane]      + cat[qBase + r * CCH + lane];
        float v1 = sY[r * CCH + lane + 32] + bo[lane + 32] + cat[qBase + r * CCH + lane + 32];
        float v2 = sY[r * CCH + lane + 64] + bo[lane + 64] + cat[qBase + r * CCH + lane + 64];
        float v3 = sY[r * CCH + lane + 96] + bo[lane + 96] + cat[qBase + r * CCH + lane + 96];
        float mu = wave_sum(v0 + v1 + v2 + v3) * (1.f / 128.f);
        float d0 = v0 - mu, d1 = v1 - mu, d2 = v2 - mu, d3 = v3 - mu;
        float var = wave_sum(d0 * d0 + d1 * d1 + d2 * d2 + d3 * d3) * (1.f / 128.f);
        float rs = rsqrtf(var + LN_EPS);
        sAcc[r * CCH + lane]      += d0 * rs * ln_g[lane]      + ln_b[lane];
        sAcc[r * CCH + lane + 32] += d1 * rs * ln_g[lane + 32] + ln_b[lane + 32];
        sAcc[r * CCH + lane + 64] += d2 * rs * ln_g[lane + 64] + ln_b[lane + 64];
        sAcc[r * CCH + lane + 96] += d3 * rs * ln_g[lane + 96] + ln_b[lane + 96];
      }
    }
  }
  __syncthreads();
  for (int i = tid; i < VV * CCH; i += 256) {
    int q = i >> 7, c = i & 127;
    att[qBase + q * CCH + c] = sAcc[i] * (1.f / 25.f);
  }
}

// ---------- stage 3: prototypes, distances, loss/acc ----------

__global__ void k_proto(const float* __restrict__ att, float* __restrict__ proto) {
  int idx = blockIdx.x * blockDim.x + threadIdx.x;
  if (idx >= NCLS * FEAT) return;
  int cl = idx / FEAT, i = idx - cl * FEAT;
  float s = 0.f;
#pragma unroll
  for (int j = 0; j < 5; ++j) s += att[(size_t)(NQRY + cl * 5 + j) * FEAT + i];
  proto[idx] = s * 0.2f;
}

__global__ void k_dist(const float* __restrict__ att, const float* __restrict__ proto,
                       float* __restrict__ dist) {
  __shared__ float red[256];
  const int n = blockIdx.x, tid = threadIdx.x;
  float d[NCLS] = {0.f, 0.f, 0.f, 0.f, 0.f};
  for (int i = tid; i < FEAT; i += 256) {
    float q = att[(size_t)n * FEAT + i];
#pragma unroll
    for (int cl = 0; cl < NCLS; ++cl) {
      float df = q - proto[(size_t)cl * FEAT + i];
      d[cl] += df * df;
    }
  }
  for (int cl = 0; cl < NCLS; ++cl) {
    red[tid] = d[cl];
    __syncthreads();
    for (int off = 128; off > 0; off >>= 1) {
      if (tid < off) red[tid] += red[tid + off];
      __syncthreads();
    }
    if (tid == 0) dist[n * NCLS + cl] = red[0];
    __syncthreads();
  }
}

__global__ void k_loss(const float* __restrict__ dist, float* __restrict__ out) {
  __shared__ float sl[128], sa[128];
  int tid = threadIdx.x;
  float lossC = 0.f, accC = 0.f;
  if (tid < NQRY) {
    float l[NCLS];
#pragma unroll
    for (int cl = 0; cl < NCLS; ++cl) l[cl] = -dist[tid * NCLS + cl];
    float mx = l[0];
#pragma unroll
    for (int cl = 1; cl < NCLS; ++cl) mx = fmaxf(mx, l[cl]);
    float se = 0.f;
#pragma unroll
    for (int cl = 0; cl < NCLS; ++cl) se += __expf(l[cl] - mx);
    float lse = __logf(se) + mx;
    int cls = tid / 15;
    lossC = -(l[cls] - lse);
    int pred = 0; float best = l[0];
#pragma unroll
    for (int cl = 1; cl < NCLS; ++cl) if (l[cl] > best) { best = l[cl]; pred = cl; }
    accC = (pred == cls) ? 1.f : 0.f;
  }
  sl[tid] = lossC; sa[tid] = accC;
  __syncthreads();
  for (int off = 64; off > 0; off >>= 1) {
    if (tid < off) { sl[tid] += sl[tid + off]; sa[tid] += sa[tid + off]; }
    __syncthreads();
  }
  if (tid == 0) { out[0] = sl[0] * (1.f / NQRY); out[1] = sa[0] * (1.f / NQRY); }
}

// ---------- host launch ----------

extern "C" void kernel_launch(void* const* d_in, const int* in_sizes, int n_in,
                              void* d_out, int out_size, void* d_ws, size_t ws_size,
                              hipStream_t stream) {
  const float* input = (const float*)d_in[0];
  const float* Wq = (const float*)d_in[1];
  const float* bq = (const float*)d_in[2];
  const float* Wk = (const float*)d_in[3];
  const float* bk = (const float*)d_in[4];
  const float* Wv = (const float*)d_in[5];
  const float* bv = (const float*)d_in[6];
  const float* Wo = (const float*)d_in[7];
  const float* bo = (const float*)d_in[8];
  const float* ln_g = (const float*)d_in[9];
  const float* ln_b = (const float*)d_in[10];
  float* out = (float*)d_out;
  (void)in_sizes; (void)n_in; (void)out_size; (void)ws_size;

  char* ws = (char*)d_ws;
  size_t off = 0;
  auto carve = [&](size_t bytes) {
    void* p = ws + off;
    off = (off + bytes + 255) & ~(size_t)255;
    return p;
  };
  u16*   Wbf   = (u16*)  carve((size_t)4 * CCH * CCH * sizeof(u16));   // swizzled weights
  float* cat   = (float*)carve((size_t)NTOT * FEAT * sizeof(float));   // residual, f32
  u16*   catBf = (u16*)  carve((size_t)NTOT * FEAT * sizeof(u16));
  u16*   Qbf   = (u16*)  carve((size_t)NTOT * FEAT * sizeof(u16));     // [n][t][v][c]
  u16*   Ktb   = (u16*)  carve((size_t)NSUP * TT * TILE_ELEMS * sizeof(u16)); // swizzled K^T
  u16*   Vbf   = (u16*)  carve((size_t)NSUP * TT * TILE_ELEMS * sizeof(u16)); // swizzled V
  float* att   = (float*)carve((size_t)NTOT * FEAT * sizeof(float));
  float* proto = (float*)carve((size_t)NCLS * FEAT * sizeof(float));
  float* dist  = (float*)carve((size_t)NQRY * NCLS * sizeof(float));

  k_weights<<<(4 * CCH * CCH + 255) / 256, 256, 0, stream>>>(Wq, Wk, Wv, Wo, Wbf);
  k_cat<<<(NTOT * FEAT) / 256, 256, 0, stream>>>(input, cat, catBf);
  k_proj<<<(NTOT * TOKS_PER + 31) / 32, 256, 0, stream>>>(
      catBf, Wbf + 0 * CCH * CCH, bq, Qbf, NTOT * TOKS_PER, 0);
  k_proj<<<(NSUP * TOKS_PER + 31) / 32, 256, 0, stream>>>(
      catBf + (size_t)NQRY * FEAT, Wbf + 1 * CCH * CCH, bk, Ktb, NSUP * TOKS_PER, 1);
  k_proj<<<(NSUP * TOKS_PER + 31) / 32, 256, 0, stream>>>(
      catBf + (size_t)NQRY * FEAT, Wbf + 2 * CCH * CCH, bv, Vbf, NSUP * TOKS_PER, 2);
  k_attn<<<NTOT * TT, 256, 0, stream>>>(Qbf, Ktb, Vbf, Wbf + 3 * CCH * CCH,
                                        cat, bo, ln_g, ln_b, att);
  k_proto<<<(NCLS * FEAT + 255) / 256, 256, 0, stream>>>(att, proto);
  k_dist<<<NQRY, 256, 0, stream>>>(att, proto, dist);
  k_loss<<<1, 128, 0, stream>>>(dist, out);
}
